// linformerAttention_8916352106604
// MI455X (gfx1250) — matrix operand order repaired and checked by reading, not compiled, not run on hardware
//
#include <hip/hip_runtime.h>

#define S_LEN   4096
#define D_DIM   64
#define DK_DIM  256
#define B_BATCH 32

typedef __attribute__((ext_vector_type(16))) __bf16         v16bf;
typedef __attribute__((ext_vector_type(8)))  float          v8f;
typedef __attribute__((ext_vector_type(8)))  unsigned short u16x8;
typedef __attribute__((ext_vector_type(4)))  float          f32x4;

union BF16Tile {
    v16bf  v;
    __bf16 b[16];
    u16x8  h[2];
};

__device__ __forceinline__ v8f wmma_bf16(const BF16Tile& a, const BF16Tile& bb, v8f c) {
    // emits v_wmma_f32_16x16x32_bf16
    return __builtin_amdgcn_wmma_f32_16x16x32_bf16(
        /*neg_a=*/false, a.v, /*neg_b=*/false, bb.v,
        /*c_mod=*/(short)0, c, /*reuse_a=*/false, /*reuse_b=*/false);
}

// -----------------------------------------------------------------------------
// Kernel 0a: streaming f32 -> bf16 convert (E_w, row-major, A-layout friendly)
// -----------------------------------------------------------------------------
__global__ __launch_bounds__(256) void cvt_bf16(const float* __restrict__ src,
                                                __bf16* __restrict__ dst) {
    size_t i = ((size_t)blockIdx.x * 256 + threadIdx.x) * 8;
    f32x4 a = *(const f32x4*)(src + i);
    f32x4 b = *(const f32x4*)(src + i + 4);
    BF16Tile t;
#pragma unroll
    for (int j = 0; j < 4; ++j) {
        t.b[j]     = (__bf16)a[j];
        t.b[4 + j] = (__bf16)b[j];
    }
    *(u16x8*)(dst + i) = t.h[0];
}

// -----------------------------------------------------------------------------
// Kernel 0b: transpose-convert x[b][s][64] f32 -> xt[b][64][S] bf16 via LDS.
// blockIdx.z selects (k -> kT) or (v -> vT).
// -----------------------------------------------------------------------------
__global__ __launch_bounds__(256) void cvt_tr(const float* __restrict__ k,
                                              const float* __restrict__ v,
                                              __bf16* __restrict__ kT,
                                              __bf16* __restrict__ vT) {
    __shared__ __bf16 tile[64][80];            // 160B rows (16B aligned)
    const float* x  = blockIdx.z ? v : k;
    __bf16*      xt = blockIdx.z ? vT : kT;
    const int b  = blockIdx.y;
    const int s0 = blockIdx.x * 64;
    const int t  = threadIdx.x;

    const int sl = t >> 2, dq = (t & 3) * 16;
    const float* xb = x + ((size_t)b * S_LEN + s0 + sl) * D_DIM + dq;
#pragma unroll
    for (int c = 0; c < 4; ++c) {
        f32x4 r = *(const f32x4*)(xb + c * 4);
#pragma unroll
        for (int j = 0; j < 4; ++j) tile[dq + c * 4 + j][sl] = (__bf16)r[j];
    }
    __syncthreads();

    const int d = t >> 2, sq = (t & 3) * 16;
    __bf16* drow = xt + ((size_t)b * D_DIM + d) * S_LEN + s0 + sq;
    *(u16x8*)(drow)     = *(const u16x8*)(&tile[d][sq]);
    *(u16x8*)(drow + 8) = *(const u16x8*)(&tile[d][sq + 8]);
}

// -----------------------------------------------------------------------------
// Phase 1: Tk = E_w @ k[b], Tv = E_w @ v[b]   (both [DK x D], reduce S=4096)
// Wave computes 16(kk) x 32(d) of BOTH projections; double-buffered K-steps.
// -----------------------------------------------------------------------------
struct P1Tiles { BF16Tile a, k0, k1, v0, v1; };

__device__ __forceinline__ void load_p1(P1Tiles& t, int s0,
    const __bf16* Arow, const __bf16* kR0, const __bf16* kR1,
    const __bf16* vR0, const __bf16* vR1, int kbase, int kb0) {
    t.a.h[0]  = *(const u16x8*)(Arow + s0 + kbase);
    t.a.h[1]  = *(const u16x8*)(Arow + s0 + kbase + 16);
    t.k0.h[0] = *(const u16x8*)(kR0 + s0 + kb0);
    t.k0.h[1] = *(const u16x8*)(kR0 + s0 + kb0 + 16);
    t.k1.h[0] = *(const u16x8*)(kR1 + s0 + kb0);
    t.k1.h[1] = *(const u16x8*)(kR1 + s0 + kb0 + 16);
    t.v0.h[0] = *(const u16x8*)(vR0 + s0 + kb0);
    t.v0.h[1] = *(const u16x8*)(vR0 + s0 + kb0 + 16);
    t.v1.h[0] = *(const u16x8*)(vR1 + s0 + kb0);
    t.v1.h[1] = *(const u16x8*)(vR1 + s0 + kb0 + 16);
}

__global__ __launch_bounds__(128) void linf_proj(
    const __bf16* __restrict__ kT, const __bf16* __restrict__ vT,
    const __bf16* __restrict__ Ewb, const float* __restrict__ Eb,
    __bf16* __restrict__ Kb, __bf16* __restrict__ Vt)
{
    const int lane  = threadIdx.x & 31;
    const int wid   = threadIdx.x >> 5;
    const int tile  = blockIdx.x * 4 + wid;     // 32 tiles = 16 kkT x 2 dT
    const int kkT   = tile >> 1;
    const int dT    = tile & 1;
    const int b     = blockIdx.y;

    const int mr    = lane & 15;
    const int kbase = (lane >> 4) * 8;
    const int kb0   = (lane >> 4) * 8;
    const int half  = (lane >> 4) * 8;

    const __bf16* Arow = Ewb + (size_t)(kkT * 16 + mr) * S_LEN;
    const __bf16* kR0  = kT + ((size_t)b * D_DIM + dT * 32 + mr)      * S_LEN;
    const __bf16* kR1  = kT + ((size_t)b * D_DIM + dT * 32 + 16 + mr) * S_LEN;
    const __bf16* vR0  = vT + ((size_t)b * D_DIM + dT * 32 + mr)      * S_LEN;
    const __bf16* vR1  = vT + ((size_t)b * D_DIM + dT * 32 + 16 + mr) * S_LEN;

    v8f accK0 = {0.f,0.f,0.f,0.f,0.f,0.f,0.f,0.f};
    v8f accK1 = accK0, accV0 = accK0, accV1 = accK0;

    P1Tiles t0, t1;
    load_p1(t0, 0, Arow, kR0, kR1, vR0, vR1, kbase, kb0);
    for (int s0 = 0; s0 < S_LEN; s0 += 64) {
        load_p1(t1, s0 + 32, Arow, kR0, kR1, vR0, vR1, kbase, kb0);
        accK0 = wmma_bf16(t0.a, t0.k0, accK0);
        accK1 = wmma_bf16(t0.a, t0.k1, accK1);
        accV0 = wmma_bf16(t0.a, t0.v0, accV0);
        accV1 = wmma_bf16(t0.a, t0.v1, accV1);
        if (s0 + 64 < S_LEN)
            load_p1(t0, s0 + 64, Arow, kR0, kR1, vR0, vR1, kbase, kb0);
        accK0 = wmma_bf16(t1.a, t1.k0, accK0);
        accK1 = wmma_bf16(t1.a, t1.k1, accK1);
        accV0 = wmma_bf16(t1.a, t1.v0, accV0);
        accV1 = wmma_bf16(t1.a, t1.v1, accV1);
    }

#pragma unroll
    for (int i = 0; i < 8; ++i) {
        const int   kk   = kkT * 16 + i + half;
        const float bias = Eb[kk];
        const int   d0   = dT * 32 + mr;
        Kb[((size_t)b * DK_DIM + kk) * D_DIM + d0]       = (__bf16)(accK0[i] + bias);
        Kb[((size_t)b * DK_DIM + kk) * D_DIM + d0 + 16]  = (__bf16)(accK1[i] + bias);
        Vt[((size_t)b * D_DIM + d0)      * DK_DIM + kk]  = (__bf16)(accV0[i] + bias);
        Vt[((size_t)b * D_DIM + d0 + 16) * DK_DIM + kk]  = (__bf16)(accV1[i] + bias);
    }
}

// -----------------------------------------------------------------------------
// Phase 2: per 16-row q tile: scores (16x256) -> softmax -> P @ V (16x64).
// -----------------------------------------------------------------------------
struct S2Tiles { BF16Tile b0, b1; };
struct O2Tiles { BF16Tile ap, bv0, bv1, bv2, bv3; };

__device__ __forceinline__ void load_s2(S2Tiles& u, int t,
    const __bf16* KbB, int mr, int kb0) {
    const __bf16* rowp = KbB + (size_t)(t * 16 + mr) * D_DIM;
    u.b0.h[0] = *(const u16x8*)(rowp + kb0);
    u.b0.h[1] = *(const u16x8*)(rowp + kb0 + 16);
    u.b1.h[0] = *(const u16x8*)(rowp + 32 + kb0);
    u.b1.h[1] = *(const u16x8*)(rowp + 32 + kb0 + 16);
}

__device__ __forceinline__ void load_o2(O2Tiles& w, int ks,
    const __bf16* Prow, const __bf16* vr0, const __bf16* vr1,
    const __bf16* vr2, const __bf16* vr3, int kbase, int kb0) {
    w.ap.h[0]  = *(const u16x8*)(Prow + ks * 32 + kbase);
    w.ap.h[1]  = *(const u16x8*)(Prow + ks * 32 + 16 + kbase);
    w.bv0.h[0] = *(const u16x8*)(vr0 + ks * 32 + kb0);
    w.bv0.h[1] = *(const u16x8*)(vr0 + ks * 32 + kb0 + 16);
    w.bv1.h[0] = *(const u16x8*)(vr1 + ks * 32 + kb0);
    w.bv1.h[1] = *(const u16x8*)(vr1 + ks * 32 + kb0 + 16);
    w.bv2.h[0] = *(const u16x8*)(vr2 + ks * 32 + kb0);
    w.bv2.h[1] = *(const u16x8*)(vr2 + ks * 32 + kb0 + 16);
    w.bv3.h[0] = *(const u16x8*)(vr3 + ks * 32 + kb0);
    w.bv3.h[1] = *(const u16x8*)(vr3 + ks * 32 + kb0 + 16);
}

__global__ __launch_bounds__(128) void linf_attn(
    const float* __restrict__ q,
    const __bf16* __restrict__ Kb,
    const __bf16* __restrict__ Vt,
    float* __restrict__ out)
{
    __shared__ __align__(16) __bf16 Pl[4][16][DK_DIM];   // 32 KB

    const int lane  = threadIdx.x & 31;
    const int wid   = threadIdx.x >> 5;
    const int b     = blockIdx.y;
    const int s0    = blockIdx.x * 64 + wid * 16;

    const int mr    = lane & 15;
    const int kbase = (lane >> 4) * 8;
    const int kb0   = (lane >> 4) * 8;
    const int half  = (lane >> 4) * 8;

    // ---- A = q tile in bf16, 1/sqrt(64)=0.125 folded in (exact pow2) ----
    const float* qrow = q + ((size_t)b * S_LEN + (s0 + mr)) * D_DIM;
    BF16Tile aq[2];
#pragma unroll
    for (int c = 0; c < 2; ++c) {
        f32x4 q0 = *(const f32x4*)(qrow + c * 32 + kbase);
        f32x4 q1 = *(const f32x4*)(qrow + c * 32 + kbase + 4);
        f32x4 q2 = *(const f32x4*)(qrow + c * 32 + kbase + 16);
        f32x4 q3 = *(const f32x4*)(qrow + c * 32 + kbase + 20);
#pragma unroll
        for (int j = 0; j < 4; ++j) {
            aq[c].b[j]      = (__bf16)(q0[j] * 0.125f);
            aq[c].b[4 + j]  = (__bf16)(q1[j] * 0.125f);
            aq[c].b[8 + j]  = (__bf16)(q2[j] * 0.125f);
            aq[c].b[12 + j] = (__bf16)(q3[j] * 0.125f);
        }
    }

    // ---- scores strip 16 x 256: double-buffered B tiles ----
    const __bf16* KbB = Kb + (size_t)b * DK_DIM * D_DIM;
    v8f acc[16];
    S2Tiles u0, u1;
    load_s2(u0, 0, KbB, mr, kb0);
#pragma unroll
    for (int t = 0; t < 16; t += 2) {
        load_s2(u1, t + 1, KbB, mr, kb0);
        v8f c0 = {0.f,0.f,0.f,0.f,0.f,0.f,0.f,0.f};
        c0 = wmma_bf16(aq[0], u0.b0, c0);
        c0 = wmma_bf16(aq[1], u0.b1, c0);
        acc[t] = c0;
        if (t + 2 < 16) load_s2(u0, t + 2, KbB, mr, kb0);
        v8f c1 = {0.f,0.f,0.f,0.f,0.f,0.f,0.f,0.f};
        c1 = wmma_bf16(aq[0], u1.b0, c1);
        c1 = wmma_bf16(aq[1], u1.b1, c1);
        acc[t + 1] = c1;
    }

    // ---- softmax over 256 per row (lanes 0-15: row i; 16-31: row i+8) ----
    float rinv[8];
#pragma unroll
    for (int i = 0; i < 8; ++i) {
        float m = -3.402823466e38f;
#pragma unroll
        for (int t = 0; t < 16; ++t) m = fmaxf(m, acc[t][i]);
#pragma unroll
        for (int off = 1; off < 16; off <<= 1)
            m = fmaxf(m, __shfl_xor(m, off, 32));   // stays within 16-lane half
        float s = 0.f;
#pragma unroll
        for (int t = 0; t < 16; ++t) {
            float e = __expf(acc[t][i] - m);
            acc[t][i] = e;
            s += e;
        }
#pragma unroll
        for (int off = 1; off < 16; off <<= 1)
            s += __shfl_xor(s, off, 32);
        rinv[i] = 1.0f / s;
    }

    // ---- C-layout -> LDS row-major [16][256] bf16 (A-layout staging) ----
#pragma unroll
    for (int t = 0; t < 16; ++t)
#pragma unroll
        for (int i = 0; i < 8; ++i)
            Pl[wid][i + half][t * 16 + mr] = (__bf16)(acc[t][i] * rinv[i]);
    __syncthreads();

    // ---- out = P @ V : ks outer so each LDS A-tile feeds 4 WMMAs ----
    const __bf16* VtB  = Vt + (size_t)b * D_DIM * DK_DIM;
    const __bf16* Prow = &Pl[wid][mr][0];
    const __bf16* vr0  = VtB + (size_t)(0 * 16 + mr) * DK_DIM;
    const __bf16* vr1  = VtB + (size_t)(1 * 16 + mr) * DK_DIM;
    const __bf16* vr2  = VtB + (size_t)(2 * 16 + mr) * DK_DIM;
    const __bf16* vr3  = VtB + (size_t)(3 * 16 + mr) * DK_DIM;

    v8f o0 = {0.f,0.f,0.f,0.f,0.f,0.f,0.f,0.f};
    v8f o1 = o0, o2 = o0, o3 = o0;
    O2Tiles w0, w1;
    load_o2(w0, 0, Prow, vr0, vr1, vr2, vr3, kbase, kb0);
#pragma unroll
    for (int ks = 0; ks < 8; ks += 2) {
        load_o2(w1, ks + 1, Prow, vr0, vr1, vr2, vr3, kbase, kb0);
        o0 = wmma_bf16(w0.ap, w0.bv0, o0);
        o1 = wmma_bf16(w0.ap, w0.bv1, o1);
        o2 = wmma_bf16(w0.ap, w0.bv2, o2);
        o3 = wmma_bf16(w0.ap, w0.bv3, o3);
        if (ks + 2 < 8) load_o2(w0, ks + 2, Prow, vr0, vr1, vr2, vr3, kbase, kb0);
        o0 = wmma_bf16(w1.ap, w1.bv0, o0);
        o1 = wmma_bf16(w1.ap, w1.bv1, o1);
        o2 = wmma_bf16(w1.ap, w1.bv2, o2);
        o3 = wmma_bf16(w1.ap, w1.bv3, o3);
    }

    float* outB = out + ((size_t)b * S_LEN + s0) * D_DIM;
#pragma unroll
    for (int i = 0; i < 8; ++i) {
        outB[(size_t)(i + half) * D_DIM +  0 + mr] = o0[i];
        outB[(size_t)(i + half) * D_DIM + 16 + mr] = o1[i];
        outB[(size_t)(i + half) * D_DIM + 32 + mr] = o2[i];
        outB[(size_t)(i + half) * D_DIM + 48 + mr] = o3[i];
    }
}

extern "C" void kernel_launch(void* const* d_in, const int* in_sizes, int n_in,
                              void* d_out, int out_size, void* d_ws, size_t ws_size,
                              hipStream_t stream) {
    const float* q  = (const float*)d_in[0];
    const float* k  = (const float*)d_in[1];
    const float* v  = (const float*)d_in[2];
    const float* Ew = (const float*)d_in[3];
    const float* Eb = (const float*)d_in[4];

    // workspace layout (bf16 elements): 2 + 16 + 16 + 1 + 1 = 36 MB
    __bf16* Ewb = (__bf16*)d_ws;
    __bf16* kT  = Ewb + (size_t)DK_DIM * S_LEN;
    __bf16* vT  = kT  + (size_t)B_BATCH * D_DIM * S_LEN;
    __bf16* Kb  = vT  + (size_t)B_BATCH * D_DIM * S_LEN;
    __bf16* Vt  = Kb  + (size_t)B_BATCH * DK_DIM * D_DIM;

    cvt_bf16<<<dim3((DK_DIM * S_LEN) / (256 * 8)), 256, 0, stream>>>(Ew, Ewb);
    cvt_tr<<<dim3(S_LEN / 64, B_BATCH, 2), 256, 0, stream>>>(k, v, kT, vT);

    linf_proj<<<dim3(8, B_BATCH), 128, 0, stream>>>(kT, vT, Ewb, Eb, Kb, Vt);
    linf_attn<<<dim3(S_LEN / 64, B_BATCH), 128, 0, stream>>>(q, Kb, Vt, (float*)d_out);
}
